// ScaledDotProductAttention_48722109006106
// MI455X (gfx1250) — matrix operand order repaired and checked by reading, not compiled, not run on hardware
//
#include <hip/hip_runtime.h>
#include <hip/hip_bf16.h>
#include <math.h>

// CDNA5 / gfx1250, wave32. WMMA 16x16x32 bf16 (f32 accumulate).
typedef __attribute__((ext_vector_type(16))) __bf16        v16bf;
typedef __attribute__((ext_vector_type(2)))  __bf16        v2bf;
typedef __attribute__((ext_vector_type(8)))  float         v8f;
typedef __attribute__((ext_vector_type(8)))  unsigned int  v8ui;

#define WAVES_PER_WG 8
// scale = 1/sqrt(64); fold log2(e) in so WMMA emits base-2 logits directly.
#define Q_PRESCALE (0.125f * 1.4426950408889634f)
#define NEG_BIG (-1e30f)

static __device__ __forceinline__ __bf16 f2bf(float x) { return (__bf16)x; }

// bare v_exp_f32: subnormal results flush to 0, which is exactly what softmax
// tails want; avoids libm's range-fixup sequence around every call.
static __device__ __forceinline__ float fast_exp2(float x) {
  return __builtin_amdgcn_exp2f(x);
}

static __device__ __forceinline__ unsigned int pack_bf16(float lo, float hi) {
  v2bf t;
  t[0] = (__bf16)lo;
  t[1] = (__bf16)hi;
  return __builtin_bit_cast(unsigned int, t);  // v_cvt_pk_bf16_f32
}

static __device__ __forceinline__ v8f wmma_bf16(v16bf a, v16bf b, v8f c) {
  // (neg_a, A, neg_b, B, c_mod, C, reuse_a, reuse_b)
  return __builtin_amdgcn_wmma_f32_16x16x32_bf16(false, a, false, b, (short)0, c,
                                                 false, false);
}

// ---------------------------------------------------------------------------
// Pre-pack kernels: lay K and V out as bf16 WMMA B-fragments (lane-ordered),
// and the mask as 1 bit per (row,col) in C-fragment lane order. One-time cost
// (~48MB traffic) removes all per-tile cvt work and scattered loads from the
// hot loops, which run 128 waves x 2 passes per head slice.
// ---------------------------------------------------------------------------

// KP layout: [bh][tile(L/16)][chunk(2)][lane(32)][8 dwords]
// dword j of a lane = bf16 pair K[tile*16 + n][c*32 + 2j + 16h (+1)], n=lane%16, h=lane/16
__global__ __launch_bounds__(256)
void pack_k_kernel(const float* __restrict__ k, unsigned int* __restrict__ KP,
                   int L, int total) {
  const int tid = blockIdx.x * 256 + threadIdx.x;
  if (tid >= total) return;
  const int j    = tid & 7;
  const int lane = (tid >> 3) & 31;
  const int c    = (tid >> 8) & 1;
  const int tile = (tid >> 9) % (L / 16);
  const int bh   = (tid >> 9) / (L / 16);
  const int n = lane & 15, h = lane >> 4;
  const int kk = c * 32 + 2 * j + 16 * h;
  const float* krow = k + ((size_t)bh * L + tile * 16 + n) * 64 + kk;
  KP[tid] = pack_bf16(krow[0], krow[1]);
}

// VP layout: [bh][tp(L/32)][t2(4)][lane(32)][8 dwords]
// dword j = bf16 pair V[tp*32 + 2j + 16h (+1)][t2*16 + n]
__global__ __launch_bounds__(256)
void pack_v_kernel(const float* __restrict__ v, unsigned int* __restrict__ VP,
                   int L, int total) {
  const int tid = blockIdx.x * 256 + threadIdx.x;
  if (tid >= total) return;
  const int j    = tid & 7;
  const int lane = (tid >> 3) & 31;
  const int t2   = (tid >> 8) & 3;
  const int tp   = (tid >> 10) % (L / 32);
  const int bh   = (tid >> 10) / (L / 32);
  const int n = lane & 15, h = lane >> 4;
  const int kk  = tp * 32 + 2 * j + 16 * h;
  const int col = t2 * 16 + n;
  const float* vb = v + (size_t)bh * L * 64;
  VP[tid] = pack_bf16(vb[(size_t)kk * 64 + col], vb[(size_t)(kk + 1) * 64 + col]);
}

// MP layout: [qt(L/16)][kt(L/16)][lane(32)] -> byte, bit r = mask[qt*16+r+8h][kt*16+n]
__global__ __launch_bounds__(256)
void pack_mask_kernel(const unsigned char* __restrict__ mask,
                      unsigned char* __restrict__ MP, int L, int total) {
  const int tid = blockIdx.x * 256 + threadIdx.x;
  if (tid >= total) return;
  const int lane = tid & 31;
  const int kt   = (tid >> 5) % (L / 16);
  const int qt   = (tid >> 5) / (L / 16);
  const int n = lane & 15, h = lane >> 4;
  unsigned int b = 0;
#pragma unroll
  for (int r = 0; r < 8; ++r)
    b |= (mask[(size_t)(qt * 16 + r + 8 * h) * L + kt * 16 + n] ? 1u : 0u) << r;
  MP[tid] = (unsigned char)b;
}

// ---------------------------------------------------------------------------
// Main attention kernel. PACKED=true uses the pre-packed fragments; false is
// the self-contained fallback (converts f32->bf16 in-loop).
// ---------------------------------------------------------------------------
template <bool PACKED>
__global__ __launch_bounds__(256)
void sdpa_fwd_attn_kernel(const float* __restrict__ q,
                          const float* __restrict__ k,
                          const float* __restrict__ v,
                          const unsigned char* __restrict__ mask,
                          const v8ui* __restrict__ KP,
                          const v8ui* __restrict__ VP,
                          const unsigned char* __restrict__ MP,
                          float* __restrict__ o_out,    // [BH, L, 64]
                          float* __restrict__ attn_out, // [BH, L, L]
                          int L) {
  constexpr int D = 64;

  // P-tile staging, bf16 16x32 per wave; pad row stride to 34 (17 dwords) to
  // spread the 16-row A-layout reads across LDS banks.
  __shared__ unsigned short Pst[WAVES_PER_WG][16][34];

  const int lane = threadIdx.x & 31;
  const int wave = threadIdx.x >> 5;
  const int half = lane >> 4;
  const int ln   = lane & 15;

  const int bh   = blockIdx.y;
  const int row0 = (blockIdx.x * WAVES_PER_WG + wave) * 16;
  const int qt   = row0 >> 4;

  const float* Qb = q + ((size_t)bh * L + row0) * D;
  const float* Kb = k + (size_t)bh * L * D;
  const float* Vb = v + (size_t)bh * L * D;
  const v8ui* KPbh = PACKED ? KP + (size_t)bh * (L / 16) * 64 : nullptr;
  const v8ui* VPbh = PACKED ? VP + (size_t)bh * (L / 32) * 128 : nullptr;
  const unsigned char* MPq = PACKED ? MP + (size_t)qt * (L / 16) * 32 : nullptr;

  // ---------------- Q -> WMMA A layout (two K=32 chunks) ----------------
  // A 16x32 operand, contiguous k order used here (not AMD's table order): lane row = ln; VGPR j holds
  // k = 2j + 16*half, pair (k, k+1) (matches packed B order).
  v16bf qa0, qa1;
  {
    const float* qrow = Qb + (size_t)ln * D;
#pragma unroll
    for (int j = 0; j < 8; ++j) {
      const int kk = 2 * j + 16 * half;
      qa0[2 * j]     = f2bf(qrow[kk] * Q_PRESCALE);
      qa0[2 * j + 1] = f2bf(qrow[kk + 1] * Q_PRESCALE);
      qa1[2 * j]     = f2bf(qrow[32 + kk] * Q_PRESCALE);
      qa1[2 * j + 1] = f2bf(qrow[32 + kk + 1] * Q_PRESCALE);
    }
  }

  // S tile for key tile base kt: 16x16 base-2 logits in C/D layout
  // (VGPR r: row = r + 8*half, col = ln).
  auto score_tile = [&](int kt) -> v8f {
    v16bf b0, b1;
    if constexpr (PACKED) {
      const int tile = kt >> 4;
      // two chunks: 2x (two global_load_b128), fully coalesced
      b0 = __builtin_bit_cast(v16bf, KPbh[(size_t)tile * 64 + lane]);
      b1 = __builtin_bit_cast(v16bf, KPbh[(size_t)tile * 64 + 32 + lane]);
      __builtin_prefetch(&KPbh[(size_t)(tile + 1) * 64 + lane], 0, 1);
    } else {
      const float* krow = Kb + (size_t)(kt + ln) * D;
      __builtin_prefetch(krow + 16 * D, 0, 1);
#pragma unroll
      for (int j = 0; j < 8; ++j) {
        const int kk = 2 * j + 16 * half;
        b0[2 * j]     = f2bf(krow[kk]);
        b0[2 * j + 1] = f2bf(krow[kk + 1]);
        b1[2 * j]     = f2bf(krow[32 + kk]);
        b1[2 * j + 1] = f2bf(krow[32 + kk + 1]);
      }
    }
    v8f s = {0.f, 0.f, 0.f, 0.f, 0.f, 0.f, 0.f, 0.f};
    s = wmma_bf16(qa0, b0, s);
    s = wmma_bf16(qa1, b1, s);
    return s;
  };

  auto mask_bits = [&](int kt) -> unsigned int {
    if constexpr (PACKED) {
      return MPq[(size_t)(kt >> 4) * 32 + lane];
    } else {
      unsigned int b = 0;
#pragma unroll
      for (int r = 0; r < 8; ++r)
        b |= (mask[(size_t)(row0 + r + 8 * half) * L + kt + ln] ? 1u : 0u) << r;
      return b;
    }
  };

  // -------- Pass 1: lane-local online softmax stats (no shuffles) --------
  float rm[8], ps[8];
#pragma unroll
  for (int r = 0; r < 8; ++r) { rm[r] = NEG_BIG; ps[r] = 0.f; }

  for (int kt = 0; kt < L; kt += 16) {
    v8f s = score_tile(kt);
    const unsigned int mb = mask_bits(kt);
#pragma unroll
    for (int r = 0; r < 8; ++r) {
      const float sc   = ((mb >> r) & 1u) ? NEG_BIG : s[r];
      // compare+selects instead of fmax/fmin: avoids v_max canonicalize ops
      const bool  snew = sc > rm[r];
      const float nm   = snew ? sc : rm[r];
      const float e    = fast_exp2((snew ? rm[r] : sc) - nm);
      ps[r] = fmaf(ps[r], snew ? e : 1.0f, snew ? 1.0f : e);
      rm[r] = nm;
    }
  }
  // Cross-lane: global row max, then rescale lane-local sums and add.
  float gm[8];
#pragma unroll
  for (int r = 0; r < 8; ++r) gm[r] = rm[r];
#pragma unroll
  for (int xm = 1; xm < 16; xm <<= 1) {
#pragma unroll
    for (int r = 0; r < 8; ++r) {
      const float o = __shfl_xor(gm[r], xm, 32);
      gm[r] = gm[r] > o ? gm[r] : o;
    }
  }
#pragma unroll
  for (int r = 0; r < 8; ++r) ps[r] *= fast_exp2(rm[r] - gm[r]);
#pragma unroll
  for (int xm = 1; xm < 16; xm <<= 1) {
#pragma unroll
    for (int r = 0; r < 8; ++r) ps[r] += __shfl_xor(ps[r], xm, 32);
  }
  float inv[8];
#pragma unroll
  for (int r = 0; r < 8; ++r) {
    inv[r] = __builtin_amdgcn_rcpf(ps[r]);  // v_rcp_f32
    rm[r]  = gm[r];
  }

  // ---------------- Pass 2: write normalized P, accumulate O = P*V ------
  float* attn_bh = attn_out + (size_t)bh * L * L;
  v8f o[4];
#pragma unroll
  for (int t2 = 0; t2 < 4; ++t2)
    o[t2] = (v8f){0.f, 0.f, 0.f, 0.f, 0.f, 0.f, 0.f, 0.f};

  for (int tp = 0; tp < L / 32; ++tp) {
#pragma unroll
    for (int sub = 0; sub < 2; ++sub) {
      const int kt = tp * 32 + sub * 16;
      v8f s = score_tile(kt);
      const unsigned int mb = mask_bits(kt);
#pragma unroll
      for (int r = 0; r < 8; ++r) {
        const int row = row0 + r + 8 * half;
        const float p =
            ((mb >> r) & 1u) ? 0.f : fast_exp2(s[r] - rm[r]) * inv[r];
        // streaming output: non-temporal so the 512MB attn write doesn't
        // evict L2-resident K/V fragments
        __builtin_nontemporal_store(p, &attn_bh[(size_t)row * L + kt + ln]);
        const __bf16 hb = f2bf(p);
        Pst[wave][r + 8 * half][sub * 16 + ln] =
            __builtin_bit_cast(unsigned short, hb);
      }
    }
    // same-wave LDS RAW: make the P-tile stores visible before A-layout reads
    asm volatile("s_wait_dscnt 0x0" ::: "memory");

    // reload P (16x32 bf16) in WMMA A layout; pairs (k,k+1) merged to b32 reads
    const unsigned int* prow = (const unsigned int*)&Pst[wave][ln][0];
    v8ui praw;
#pragma unroll
    for (int j = 0; j < 8; ++j) {
      const int dw = j + 8 * half;
      praw[j] = prow[dw];
    }
    const v16bf pa = __builtin_bit_cast(v16bf, praw);

    // O[:, t2*16:+16] += P(16x32) * V(32x16)
#pragma unroll
    for (int t2 = 0; t2 < 4; ++t2) {
      v16bf b;
      if constexpr (PACKED) {
        b = __builtin_bit_cast(v16bf, VPbh[(size_t)tp * 128 + t2 * 32 + lane]);
      } else {
        const int kb = tp * 32;
#pragma unroll
        for (int j = 0; j < 8; ++j) {
          const int kk = 2 * j + 16 * half;
          b[2 * j]     = f2bf(Vb[(size_t)(kb + kk) * D + t2 * 16 + ln]);
          b[2 * j + 1] = f2bf(Vb[(size_t)(kb + kk + 1) * D + t2 * 16 + ln]);
        }
      }
      o[t2] = wmma_bf16(pa, b, o[t2]);
    }
  }

  // P was already normalized, so O is final.
#pragma unroll
  for (int t2 = 0; t2 < 4; ++t2) {
#pragma unroll
    for (int r = 0; r < 8; ++r) {
      const int row = row0 + r + 8 * half;
      __builtin_nontemporal_store(
          o[t2][r], &o_out[((size_t)bh * L + row) * D + t2 * 16 + ln]);
    }
  }
}

extern "C" void kernel_launch(void* const* d_in, const int* in_sizes, int n_in,
                              void* d_out, int out_size, void* d_ws,
                              size_t ws_size, hipStream_t stream) {
  const float* q = (const float*)d_in[0];
  const float* k = (const float*)d_in[1];
  const float* v = (const float*)d_in[2];
  const unsigned char* mask = (const unsigned char*)d_in[3]; // bool [1,1,L,L]

  constexpr int D = 64; // SCALE_D in the reference
  int L = 1;
  while ((long long)L * L < (long long)in_sizes[3]) L <<= 1;
  const int BH = in_sizes[0] / (L * D);

  float* out      = (float*)d_out;
  float* o_out    = out;                      // [BH, L, D]
  float* attn_out = out + (size_t)BH * L * D; // [BH, L, L]

  const dim3 grid(L / (16 * WAVES_PER_WG), BH);

  const size_t nKP = (size_t)BH * L * 32;              // dwords
  const size_t nVP = (size_t)BH * L * 32;              // dwords
  const size_t nMP = (size_t)(L / 16) * (L / 16) * 32; // bytes
  const size_t need = nKP * 4 + nVP * 4 + nMP;

  if (d_ws != nullptr && ws_size >= need) {
    unsigned int*  KP = (unsigned int*)d_ws;
    unsigned int*  VP = KP + nKP;
    unsigned char* MP = (unsigned char*)(VP + nVP);
    pack_k_kernel<<<(int)((nKP + 255) / 256), 256, 0, stream>>>(k, KP, L, (int)nKP);
    pack_v_kernel<<<(int)((nVP + 255) / 256), 256, 0, stream>>>(v, VP, L, (int)nVP);
    pack_mask_kernel<<<(int)((nMP + 255) / 256), 256, 0, stream>>>(mask, MP, L, (int)nMP);
    sdpa_fwd_attn_kernel<true><<<grid, 32 * WAVES_PER_WG, 0, stream>>>(
        q, k, v, mask, (const v8ui*)KP, (const v8ui*)VP, MP, o_out, attn_out, L);
  } else {
    sdpa_fwd_attn_kernel<false><<<grid, 32 * WAVES_PER_WG, 0, stream>>>(
        q, k, v, mask, nullptr, nullptr, nullptr, o_out, attn_out, L);
  }
}
